// EasierLMAttention_44994077393193
// MI455X (gfx1250) — matrix operand order repaired and checked by reading, not compiled, not run on hardware
//
#include <hip/hip_runtime.h>
#include <hip/hip_bf16.h>

// ---------------------------------------------------------------------------
// MI455X (gfx1250) GQA attention block: bf16 WMMA + async global->LDS copies.
//   hidden (2,2048,2048) f32 -> out (2,2048,2048) f32
// ---------------------------------------------------------------------------

typedef __bf16 bf16;
typedef __attribute__((ext_vector_type(8)))  bf16  v8bf;
typedef __attribute__((ext_vector_type(16))) bf16  v16bf;
typedef __attribute__((ext_vector_type(8)))  float v8f;

#define D_MODEL 2048
#define NH      32
#define NKV     8
#define HD      64
#define SEQ     2048
#define BATCH   2
#define NTOK    (BATCH*SEQ)          // 4096 rows
#define NQKV    (D_MODEL + 2*NKV*HD) // 3072 fused QKV columns
#define THETA_F 10000.0f

static __device__ __forceinline__ v8f wmma_bf16(v16bf a, v16bf b, v8f c) {
  // D = A(16x32 bf16) x B(32x16 bf16) + C(16x16 f32)
  return __builtin_amdgcn_wmma_f32_16x16x32_bf16(false, a, false, b,
                                                 (short)0, c, false, false);
}

// Build a 16-element bf16 fragment from two contiguous 8-element (16B) chunks.
static __device__ __forceinline__ v16bf joinfrag(const bf16* p0, const bf16* p1) {
  v8bf lo = *(const v8bf*)p0;
  v8bf hi = *(const v8bf*)p1;
  v16bf r;
#pragma unroll
  for (int i = 0; i < 8; ++i) { r[i] = lo[i]; r[i + 8] = hi[i]; }
  return r;
}

// CDNA5 async copy: 16B global -> LDS per lane, tracked by ASYNCcnt.
static __device__ __forceinline__ void async_load16(const bf16* g, unsigned ldsOff) {
  asm volatile("global_load_async_to_lds_b128 %0, %1, off"
               :: "v"(ldsOff), "v"((unsigned long long)(const void*)g)
               : "memory");
}
static __device__ __forceinline__ void wait_async0() {
  asm volatile("s_wait_asynccnt 0x0" ::: "memory");
}
// Low 32 bits of a generic pointer to __shared__ == LDS byte offset.
static __device__ __forceinline__ unsigned lds_off(const void* p) {
  return (unsigned)(unsigned long long)p;
}

// ---------------------------------------------------------------------------
// Elementwise f32 -> bf16 convert
// ---------------------------------------------------------------------------
__global__ void cvt_f32_bf16(const float* __restrict__ in, bf16* __restrict__ out, int n) {
  int i = blockIdx.x * blockDim.x + threadIdx.x;
  if (i < n) out[i] = (bf16)in[i];
}

// W (Kd x Nd, row-major f32) -> Wt (Nd x Kd, row-major bf16) "K-contiguous"
__global__ void cvt_transpose_bf16(const float* __restrict__ W, bf16* __restrict__ Wt,
                                   int Kd, int Nd) {
  int i = blockIdx.x * blockDim.x + threadIdx.x;
  if (i >= Kd * Nd) return;
  int k = i / Nd, n = i - k * Nd;
  Wt[(size_t)n * Kd + k] = (bf16)W[i];
}

// ---------------------------------------------------------------------------
// NT GEMM: C(MxN f32) = A(MxK bf16, row-major) * Bt(NxK bf16, row-major)^T
// Block tile 128x128, BK=32, 8 waves each owning a 32x64 region (8 WMMA/step).
// Tiles stream in via GLOBAL_LOAD_ASYNC_TO_LDS_B128 pipelined one step ahead:
//   wait_async -> barrier -> ds_load frags -> barrier(dscnt0) -> issue next
//   -> 8x WMMA (next tile in flight during the matrix ops).
// Requires M%128==0, N%128==0, K%32==0 (true for all uses here).
// ---------------------------------------------------------------------------
__global__ __launch_bounds__(256) void gemm_nt_bf16(const bf16* __restrict__ A,
                                                    const bf16* __restrict__ Bt,
                                                    float* __restrict__ C,
                                                    int M, int N, int K) {
  __shared__ __align__(16) bf16 lA[128 * 40];  // 32 bf16 per row + 8 pad (80B stride)
  __shared__ __align__(16) bf16 lB[128 * 40];
  const int tid  = threadIdx.x;
  const int lane = tid & 31, wid = tid >> 5;
  const int wm = wid & 3, wn = wid >> 2;       // 4x2 wave grid
  const int m0 = blockIdx.y * 128, n0 = blockIdx.x * 128;
  const int r16 = lane & 15, kh = lane >> 4;
  (void)M;

  // per-thread copy slots: rows sr and sr+64, 16B quad sq (512 slots / 256 thr)
  const int sr = tid >> 2, sq = (tid & 3) * 8;
  const unsigned laOff0 = lds_off(&lA[sr * 40 + sq]);
  const unsigned laOff1 = lds_off(&lA[(sr + 64) * 40 + sq]);
  const unsigned lbOff0 = lds_off(&lB[sr * 40 + sq]);
  const unsigned lbOff1 = lds_off(&lB[(sr + 64) * 40 + sq]);
  const bf16* gA0 = A  + (size_t)(m0 + sr) * K + sq;
  const bf16* gA1 = A  + (size_t)(m0 + sr + 64) * K + sq;
  const bf16* gB0 = Bt + (size_t)(n0 + sr) * K + sq;
  const bf16* gB1 = Bt + (size_t)(n0 + sr + 64) * K + sq;

  v8f acc[2][4] = {};

  // prologue: first tile in flight
  async_load16(gA0, laOff0); async_load16(gA1, laOff1);
  async_load16(gB0, lbOff0); async_load16(gB1, lbOff1);

  for (int kt = 0; kt < K; kt += 32) {
    wait_async0();          // this wave's slots landed
    __syncthreads();        // everyone's slots landed

    v16bf af[2], bfr[4];
#pragma unroll
    for (int mt = 0; mt < 2; ++mt) {
      int r = wm * 32 + mt * 16 + r16;
      af[mt] = joinfrag(&lA[r * 40 + kh * 8], &lA[r * 40 + 16 + kh * 8]);
    }
#pragma unroll
    for (int nt = 0; nt < 4; ++nt) {
      int r = wn * 64 + nt * 16 + r16;
      bfr[nt] = joinfrag(&lB[r * 40 + kh * 8], &lB[r * 40 + 16 + kh * 8]);
    }
    __syncthreads();        // barrier fence waits DScnt==0: frags are in regs

    if (kt + 32 < K) {      // stream next tile while WMMAs run
      int k2 = kt + 32;
      async_load16(gA0 + k2, laOff0); async_load16(gA1 + k2, laOff1);
      async_load16(gB0 + k2, lbOff0); async_load16(gB1 + k2, lbOff1);
    }
#pragma unroll
    for (int mt = 0; mt < 2; ++mt)
#pragma unroll
      for (int nt = 0; nt < 4; ++nt)
        acc[mt][nt] = wmma_bf16(af[mt], bfr[nt], acc[mt][nt]);
  }

  // epilogue: 64-bit base once, small int offsets per store
  float* Cw = C + (size_t)(m0 + wm * 32 + 8 * kh) * N + (n0 + wn * 64 + r16);
#pragma unroll
  for (int mt = 0; mt < 2; ++mt)
#pragma unroll
    for (int nt = 0; nt < 4; ++nt)
#pragma unroll
      for (int v = 0; v < 8; ++v)
        Cw[(mt * 16 + v) * N + nt * 16] = acc[mt][nt][v];
}

// ---------------------------------------------------------------------------
// RoPE (interleaved pairs) + repack into attention-friendly bf16 layouts:
//   Qb[b][h][s][64], Kb[b][hk][s][64], Vt[b][hk][64][s]
// ---------------------------------------------------------------------------
__global__ void rope_pack(const float* __restrict__ qkv, const int* __restrict__ pos_ids,
                          bf16* __restrict__ Qb, bf16* __restrict__ Kb,
                          bf16* __restrict__ Vt) {
  int idx = blockIdx.x * blockDim.x + threadIdx.x;
  if (idx >= NTOK * (NQKV / 2)) return;
  int t  = idx / (NQKV / 2);
  int c2 = idx - t * (NQKV / 2);
  int b  = t / SEQ, sp = t - b * SEQ;
  const float* row = qkv + (size_t)t * NQKV;
  int col = c2 * 2;

  if (col < D_MODEL + NKV * HD) {              // Q or K: rotate pair
    float x0 = row[col], x1 = row[col + 1];
    int isK = col >= D_MODEL;
    int cc  = isK ? (col - D_MODEL) : col;
    int hh  = cc / HD, d = cc % HD;            // d even
    float pos = (float)pos_ids[b * SEQ + sp];
    float ang = pos * __powf(THETA_F, -(float)d / (float)HD);
    float sn, cs; __sincosf(ang, &sn, &cs);
    float o0 = x0 * cs - x1 * sn;
    float o1 = x0 * sn + x1 * cs;
    bf16* dst = isK ? (Kb + (((size_t)(b * NKV + hh)) * SEQ + sp) * HD + d)
                    : (Qb + (((size_t)(b * NH  + hh)) * SEQ + sp) * HD + d);
    dst[0] = (bf16)o0; dst[1] = (bf16)o1;
  } else {                                     // V: transpose into [d][s]
    int cc = col - (D_MODEL + NKV * HD);
    int hh = cc / HD, d = cc % HD;
    bf16* base = Vt + ((size_t)(b * NKV + hh)) * HD * SEQ;
    base[(size_t)d       * SEQ + sp] = (bf16)row[col];
    base[(size_t)(d + 1) * SEQ + sp] = (bf16)row[col + 1];
  }
}

// ---------------------------------------------------------------------------
// Flash attention (causal, GQA g=4). One block = (b, h, 128 q rows); 8 waves,
// each owns a 16-row q tile. K/V stream through double-buffered LDS via async
// copies: next 32-key tile is in flight while the current one is computed.
// ---------------------------------------------------------------------------
__global__ __launch_bounds__(256) void attn_fwd(const bf16* __restrict__ Qb,
                                                const bf16* __restrict__ Kb,
                                                const bf16* __restrict__ Vt,
                                                const int*  __restrict__ amask,
                                                bf16* __restrict__ Ob) {
  __shared__ __align__(16) bf16 lK[2][32 * 72];   // 32 keys x 64d (+pad, 144B rows)
  __shared__ __align__(16) bf16 lV[2][64 * 40];   // 64 d   x 32 keys (+pad, 80B rows)
  __shared__ __align__(16) bf16 lP[8 * 16 * 40];  // per-wave 16x32 P scratch

  const int tid = threadIdx.x, lane = tid & 31, wid = tid >> 5;
  const int r16 = lane & 15, kh = lane >> 4;
  const int bid  = blockIdx.x;
  const int qblk = bid & 15;
  const int h    = (bid >> 4) & 31;
  const int b    = bid >> 9;
  const int hk   = h >> 2;                       // GQA: 4 Q heads per KV head
  const int q0   = qblk * 128;
  const int qbase = q0 + wid * 16;

  // Q fragments for this wave's 16 rows (2 k-steps over d=64)
  const bf16* qptr = Qb + (((size_t)(b * NH + h)) * SEQ + qbase + r16) * HD;
  v16bf qf[2];
#pragma unroll
  for (int s = 0; s < 2; ++s)
    qf[s] = joinfrag(qptr + s * 32 + kh * 8, qptr + s * 32 + 16 + kh * 8);

  v8f O[4] = {};
  float rm[8], rl[8];
#pragma unroll
  for (int v = 0; v < 8; ++v) { rm[v] = -1e30f; rl[v] = 0.0f; }

  const bf16* kbase = Kb + ((size_t)(b * NKV + hk)) * SEQ * HD;
  const bf16* vbase = Vt + ((size_t)(b * NKV + hk)) * HD * SEQ;
  const int nkt = (qblk + 1) * 4;                // causal: keys <= q0+127

  // per-thread async copy slots (one 16B K slot + one 16B V slot per tile)
  const int kr = tid >> 3, kc = (tid & 7) * 8;   // K: 32 rows x 8 quads
  const int vr = tid >> 2, vc = (tid & 3) * 8;   // V: 64 rows x 4 quads
  const unsigned lkOff[2] = { lds_off(&lK[0][kr * 72 + kc]), lds_off(&lK[1][kr * 72 + kc]) };
  const unsigned lvOff[2] = { lds_off(&lV[0][vr * 40 + vc]), lds_off(&lV[1][vr * 40 + vc]) };
  const bf16* gK = kbase + (size_t)kr * HD + kc;   // + kt*HD per tile
  const bf16* gV = vbase + (size_t)vr * SEQ + vc;  // + kt    per tile

  // prologue: tile 0 -> buffer 0
  async_load16(gK, lkOff[0]);
  async_load16(gV, lvOff[0]);

  for (int kti = 0; kti < nkt; ++kti) {
    const int kt  = kti * 32;
    const int cur = kti & 1;

    wait_async0();          // current tile landed (this wave's slots)
    __syncthreads();        // all waves' slots landed; prev reads done (tail barrier)

    if (kti + 1 < nkt) {    // stream next tile into the other buffer
      int kn = kt + 32;
      async_load16(gK + (size_t)kn * HD, lkOff[cur ^ 1]);
      async_load16(gV + kn,              lvOff[cur ^ 1]);
    }

    if (kt <= qbase + 15) {                      // wave-uniform causal skip
      const bf16* Kc = lK[cur];
      const bf16* Vc = lV[cur];

      // S = Q K^T (two 16x16 key tiles, contraction d=64 in 2 WMMA steps)
      v8f s0 = {}, s1 = {};
#pragma unroll
      for (int st = 0; st < 2; ++st) {
        v16bf k0 = joinfrag(&Kc[r16 * 72 + st * 32 + kh * 8],
                            &Kc[r16 * 72 + st * 32 + 16 + kh * 8]);
        v16bf k1 = joinfrag(&Kc[(16 + r16) * 72 + st * 32 + kh * 8],
                            &Kc[(16 + r16) * 72 + st * 32 + 16 + kh * 8]);
        s0 = wmma_bf16(qf[st], k0, s0);
        s1 = wmma_bf16(qf[st], k1, s1);
      }

      // scale + causal/attention mask (lane owns one key column)
      const int  kg0 = kt + r16, kg1 = kt + 16 + r16;
      const bool mk0 = amask[b * SEQ + kg0] > 0;
      const bool mk1 = amask[b * SEQ + kg1] > 0;
#pragma unroll
      for (int v = 0; v < 8; ++v) {
        int qg = qbase + v + 8 * kh;
        float a0 = s0[v] * 0.125f, a1 = s1[v] * 0.125f;
        s0[v] = (mk0 && kg0 <= qg) ? a0 : -1e30f;
        s1[v] = (mk1 && kg1 <= qg) ? a1 : -1e30f;
      }

      // online softmax: row stats via width-16 shuffles (rows live per half-wave)
#pragma unroll
      for (int v = 0; v < 8; ++v) {
        float mx = fmaxf(s0[v], s1[v]);
#pragma unroll
        for (int o = 8; o >= 1; o >>= 1) mx = fmaxf(mx, __shfl_xor(mx, o, 16));
        float mn    = fmaxf(rm[v], mx);
        float alpha = __expf(rm[v] - mn);
        rm[v] = mn;
        float p0 = __expf(s0[v] - mn);
        float p1 = __expf(s1[v] - mn);
        s0[v] = p0; s1[v] = p1;
        float rs = p0 + p1;
#pragma unroll
        for (int o = 8; o >= 1; o >>= 1) rs += __shfl_xor(rs, o, 16);
        rl[v] = rl[v] * alpha + rs;
#pragma unroll
        for (int nd = 0; nd < 4; ++nd) O[nd][v] *= alpha;
      }

      // P (C-layout) -> per-wave LDS scratch -> A-fragment layout
      bf16* pw = &lP[wid * 640];
#pragma unroll
      for (int v = 0; v < 8; ++v) {
        int m = v + 8 * kh;
        pw[m * 40 + r16]      = (bf16)s0[v];
        pw[m * 40 + 16 + r16] = (bf16)s1[v];
      }
      v16bf pf = joinfrag(&pw[r16 * 40 + kh * 8], &pw[r16 * 40 + 16 + kh * 8]);

      // O += P * V (4 d-tiles)
#pragma unroll
      for (int nd = 0; nd < 4; ++nd) {
        int d = nd * 16 + r16;
        v16bf vf = joinfrag(&Vc[d * 40 + kh * 8], &Vc[d * 40 + 16 + kh * 8]);
        O[nd] = wmma_bf16(pf, vf, O[nd]);
      }
    }
    __syncthreads();        // fence (DScnt==0): all reads of `cur` finished
  }

  // normalize and emit bf16 for the Wo GEMM
#pragma unroll
  for (int nd = 0; nd < 4; ++nd) {
    int d = nd * 16 + r16;
#pragma unroll
    for (int v = 0; v < 8; ++v) {
      int qg = qbase + v + 8 * kh;
      float o = O[nd][v] / fmaxf(rl[v], 1e-20f);
      Ob[((size_t)(b * SEQ) + qg) * D_MODEL + h * HD + d] = (bf16)o;
    }
  }
}

// ---------------------------------------------------------------------------
// Launcher
// ---------------------------------------------------------------------------
extern "C" void kernel_launch(void* const* d_in, const int* in_sizes, int n_in,
                              void* d_out, int out_size, void* d_ws, size_t ws_size,
                              hipStream_t stream) {
  (void)in_sizes; (void)n_in; (void)out_size; (void)ws_size;
  const float* hs    = (const float*)d_in[0];
  const int*   amask = (const int*)d_in[1];
  const int*   pos   = (const int*)d_in[2];
  const float* Wq    = (const float*)d_in[3];
  const float* Wk    = (const float*)d_in[4];
  const float* Wv    = (const float*)d_in[5];
  const float* Wo    = (const float*)d_in[6];
  float* out = (float*)d_out;

  // workspace carve (bytes)
  char* ws = (char*)d_ws;
  bf16*  Xb    = (bf16*)(ws);                              // 4096x2048      16.8MB
  bf16*  WtQKV = (bf16*)(ws + 16777216);                   // 3072x2048      12.6MB
  bf16*  WtO   = (bf16*)(ws + 29360128);                   // 2048x2048       8.4MB
  float* QKVf  = (float*)(ws + 37748736);                  // 4096x3072 f32  50.3MB
  bf16*  Qb    = (bf16*)(ws + 88080384);                   // 2x32x2048x64   16.8MB
  bf16*  Kb    = (bf16*)(ws + 104857600);                  // 2x8x2048x64     4.2MB
  bf16*  Vt    = (bf16*)(ws + 109051904);                  // 2x8x64x2048     4.2MB
  bf16*  Ab    = (bf16*)(ws + 113246208);                  // 4096x2048      16.8MB

  const int thr = 256;

  // 1) bf16 conversions (+ weight transposes into K-contiguous layout)
  cvt_f32_bf16<<<(NTOK * D_MODEL + thr - 1) / thr, thr, 0, stream>>>(hs, Xb, NTOK * D_MODEL);
  cvt_transpose_bf16<<<(D_MODEL * D_MODEL + thr - 1) / thr, thr, 0, stream>>>(
      Wq, WtQKV, D_MODEL, D_MODEL);                                   // rows 0..2047
  cvt_transpose_bf16<<<(D_MODEL * NKV * HD + thr - 1) / thr, thr, 0, stream>>>(
      Wk, WtQKV + (size_t)D_MODEL * D_MODEL, D_MODEL, NKV * HD);      // rows 2048..2559
  cvt_transpose_bf16<<<(D_MODEL * NKV * HD + thr - 1) / thr, thr, 0, stream>>>(
      Wv, WtQKV + (size_t)(D_MODEL + NKV * HD) * D_MODEL, D_MODEL, NKV * HD); // 2560..3071
  cvt_transpose_bf16<<<(D_MODEL * D_MODEL + thr - 1) / thr, thr, 0, stream>>>(
      Wo, WtO, D_MODEL, D_MODEL);

  // 2) fused QKV projection: (4096x2048) x (2048x3072) -> f32
  gemm_nt_bf16<<<dim3(NQKV / 128, NTOK / 128), thr, 0, stream>>>(
      Xb, WtQKV, QKVf, NTOK, NQKV, D_MODEL);

  // 3) RoPE + repack to Qb/Kb/Vt bf16
  rope_pack<<<(NTOK * (NQKV / 2) + thr - 1) / thr, thr, 0, stream>>>(
      QKVf, pos, Qb, Kb, Vt);

  // 4) flash attention -> Ab bf16
  attn_fwd<<<BATCH * NH * (SEQ / 128), thr, 0, stream>>>(Qb, Kb, Vt, amask, Ab);

  // 5) output projection: (4096x2048) x (2048x2048) -> f32 out
  gemm_nt_bf16<<<dim3(D_MODEL / 128, NTOK / 128), thr, 0, stream>>>(
      Ab, WtO, out, NTOK, D_MODEL, D_MODEL);
}
